// PointNet2_40286793236952
// MI455X (gfx1250) — matrix operand order repaired and checked by reading, not compiled, not run on hardware
//
#include <hip/hip_runtime.h>

// ============================================================================
// PointNet++ (single cloud, N=16384) for gfx1250 / MI455X.
//   - FPS: one persistent 1024-thread WG, register dists, LDS argmax reduce.
//   - Ball query (top-32) / kNN(3): register-resident insertion top-k.
//   - All MLPs: wave32 WMMA v_wmma_f32_16x16x32_f16, f16 in / f32 accumulate.
//     32-row tiles per 128-thread WG, f16 activations staged in LDS,
//     final layer kept in registers -> direct store or masked max-aggregate.
// ============================================================================

typedef __attribute__((ext_vector_type(16))) _Float16 v16h;
typedef __attribute__((ext_vector_type(8)))  _Float16 v8h;
typedef __attribute__((ext_vector_type(8)))  float    v8f;

union V16U { v16h v; v8h h[2]; };

// ------------------------------ FPS -----------------------------------------
__global__ void __launch_bounds__(1024)
fps_kernel(const float* __restrict__ pos, int N, int nOut, int* __restrict__ outIdx) {
  const int tid = threadIdx.x;
  __shared__ float sBestD[32];
  __shared__ int   sBestI[32];
  __shared__ int   sLast;
  float d[16];
#pragma unroll
  for (int i = 0; i < 16; ++i) d[i] = 1e30f;
  if (tid == 0) { outIdx[0] = 0; sLast = 0; }
  __syncthreads();
  for (int s = 1; s < nOut; ++s) {
    const int last = sLast;
    const float lx = pos[last * 3 + 0], ly = pos[last * 3 + 1], lz = pos[last * 3 + 2];
    float bd = -1.0f; int bi = 0x7fffffff;
#pragma unroll
    for (int i = 0; i < 16; ++i) {
      int p = tid + (i << 10);
      if (p < N) {
        float dx = pos[p * 3 + 0] - lx;
        float dy = pos[p * 3 + 1] - ly;
        float dz = pos[p * 3 + 2] - lz;
        float dd = dx * dx + dy * dy + dz * dz;
        float nd = fminf(d[i], dd);
        d[i] = nd;
        if (nd > bd) { bd = nd; bi = p; }   // later p is larger -> ties keep low idx
      }
    }
    // wave32 reduce (max dist, min index tiebreak)
#pragma unroll
    for (int off = 16; off > 0; off >>= 1) {
      float od = __shfl_down(bd, off);
      int   oi = __shfl_down(bi, off);
      if (od > bd || (od == bd && oi < bi)) { bd = od; bi = oi; }
    }
    int wid = tid >> 5;
    if ((tid & 31) == 0) { sBestD[wid] = bd; sBestI[wid] = bi; }
    __syncthreads();
    if (tid == 0) {
      float BB = sBestD[0]; int BI = sBestI[0];
      for (int w = 1; w < 32; ++w) {
        float od = sBestD[w]; int oi = sBestI[w];
        if (od > BB || (od == BB && oi < BI)) { BB = od; BI = oi; }
      }
      sLast = BI; outIdx[s] = BI;
    }
    __syncthreads();
  }
}

// ------------------------------ gather --------------------------------------
__global__ void gather_pos_kernel(const float* __restrict__ pos, const int* __restrict__ idx,
                                  int Q, float* __restrict__ posQ) {
  int i = blockIdx.x * blockDim.x + threadIdx.x;
  if (i < Q * 3) posQ[i] = pos[idx[i / 3] * 3 + (i % 3)];
}

// ------------------------------ ball query (top-32) -------------------------
__global__ void __launch_bounds__(128)
ball_query_kernel(const float* __restrict__ posQ, int Q, const float* __restrict__ posS, int Ns,
                  float r2, int* __restrict__ nbr, unsigned* __restrict__ mask) {
  int q = blockIdx.x * blockDim.x + threadIdx.x;
  if (q >= Q) return;
  const float qx = posQ[q * 3 + 0], qy = posQ[q * 3 + 1], qz = posQ[q * 3 + 2];
  const float qn = qx * qx + qy * qy + qz * qz;
  float dist[32]; int idx[32];
#pragma unroll
  for (int j = 0; j < 32; ++j) { dist[j] = 3.4e38f; idx[j] = 0; }
  for (int s = 0; s < Ns; ++s) {
    float sx = posS[s * 3 + 0], sy = posS[s * 3 + 1], sz = posS[s * 3 + 2];
    float dd = qn + sx * sx + sy * sy + sz * sz - 2.0f * (qx * sx + qy * sy + qz * sz);
    if (dd < dist[31]) {
      float cd = dd; int ci = s;
#pragma unroll
      for (int j = 0; j < 32; ++j) {   // stable bubble insert: strict '<' keeps low idx on ties
        if (cd < dist[j]) {
          float td = dist[j]; dist[j] = cd; cd = td;
          int   ti = idx[j];  idx[j] = ci;  ci = ti;
        }
      }
    }
  }
  unsigned m = 0u;
#pragma unroll
  for (int j = 0; j < 32; ++j) {
    nbr[q * 32 + j] = idx[j];
    if (dist[j] <= r2) m |= (1u << j);
  }
  mask[q] = m;
}

// ------------------------------ kNN (k=3) -----------------------------------
__global__ void __launch_bounds__(128)
knn3_kernel(const float* __restrict__ posD, int Nd, const float* __restrict__ posS, int Ns,
            int* __restrict__ idx, float* __restrict__ w) {
  int n = blockIdx.x * blockDim.x + threadIdx.x;
  if (n >= Nd) return;
  const float qx = posD[n * 3 + 0], qy = posD[n * 3 + 1], qz = posD[n * 3 + 2];
  const float qn = qx * qx + qy * qy + qz * qz;
  float d0 = 3.4e38f, d1 = 3.4e38f, d2 = 3.4e38f;
  int i0 = 0, i1 = 0, i2 = 0;
  for (int s = 0; s < Ns; ++s) {
    float sx = posS[s * 3 + 0], sy = posS[s * 3 + 1], sz = posS[s * 3 + 2];
    float dd = qn + sx * sx + sy * sy + sz * sz - 2.0f * (qx * sx + qy * sy + qz * sz);
    if (dd < d2) {
      if (dd < d0)      { d2 = d1; i2 = i1; d1 = d0; i1 = i0; d0 = dd; i0 = s; }
      else if (dd < d1) { d2 = d1; i2 = i1; d1 = dd; i1 = s; }
      else              { d2 = dd; i2 = s; }
    }
  }
  idx[n * 3 + 0] = i0; idx[n * 3 + 1] = i1; idx[n * 3 + 2] = i2;
  w[n * 3 + 0] = 1.0f / fmaxf(d0, 1e-16f);
  w[n * 3 + 1] = 1.0f / fmaxf(d1, 1e-16f);
  w[n * 3 + 2] = 1.0f / fmaxf(d2, 1e-16f);
}

// ------------------------------ interpolation -------------------------------
__global__ void interp_kernel(const float* __restrict__ xS, int F, const int* __restrict__ idx,
                              const float* __restrict__ w, int Nd, float* __restrict__ out) {
  int i = blockIdx.x * blockDim.x + threadIdx.x;
  if (i >= Nd * F) return;
  int n = i / F, f = i % F;
  float w0 = w[n * 3 + 0], w1 = w[n * 3 + 1], w2 = w[n * 3 + 2];
  float ws = w0 + w1 + w2;
  float v = w0 * xS[(long long)idx[n * 3 + 0] * F + f]
          + w1 * xS[(long long)idx[n * 3 + 1] * F + f]
          + w2 * xS[(long long)idx[n * 3 + 2] * F + f];
  out[i] = v / ws;
}

// --------------------- build SA edge-input matrix (f16) ---------------------
__global__ void build_sa_input_kernel(const float* __restrict__ x, int F,
                                      const float* __restrict__ posS, const float* __restrict__ posQ,
                                      const int* __restrict__ nbr, int E, int kpad,
                                      _Float16* __restrict__ out) {
  long long i = (long long)blockIdx.x * blockDim.x + threadIdx.x;
  if (i >= (long long)E * kpad) return;
  int r = (int)(i / kpad), c = (int)(i % kpad);
  int q = r >> 5;
  int s = nbr[r];
  float v = 0.0f;
  if (c < F)           v = x[(long long)s * F + c];
  else if (c < F + 3)  v = posS[s * 3 + (c - F)] - posQ[q * 3 + (c - F)];
  out[i] = (_Float16)v;
}

// --------------------- build FP concat input matrix (f16) -------------------
__global__ void build_fp_input_kernel(const float* __restrict__ h, int F1,
                                      const float* __restrict__ skip, int F2,
                                      int Nd, int kpad, _Float16* __restrict__ out) {
  long long i = (long long)blockIdx.x * blockDim.x + threadIdx.x;
  if (i >= (long long)Nd * kpad) return;
  int n = (int)(i / kpad), c = (int)(i % kpad);
  float v = 0.0f;
  if (c < F1)            v = h[(long long)n * F1 + c];
  else if (c < F1 + F2)  v = skip[(long long)n * F2 + (c - F1)];
  out[i] = (_Float16)v;
}

// --------------------- weight prep: f32 [K][H] -> f16 [Hpad][Kpad] ----------
__global__ void prep_weight_kernel(const float* __restrict__ W, const float* __restrict__ b,
                                   int K, int H, int kpad, int hpad,
                                   _Float16* __restrict__ WT, float* __restrict__ bp) {
  int i = blockIdx.x * blockDim.x + threadIdx.x;
  if (i < hpad * kpad) {
    int o = i / kpad, k = i % kpad;
    WT[i] = (o < H && k < K) ? (_Float16)W[(long long)k * H + o] : (_Float16)0.0f;
  }
  if (i < hpad) bp[i] = (i < H) ? b[i] : 0.0f;
}

// --------------------- WMMA MLP (up to 3 layers, opt. max-aggregate) --------
// A frag (16x32 f16, M across lanes): lane l: row=l&15, h=l>>4,
//   k = kb + (e<8 ? h*8+e : 16+h*8+(e-8))   -> two contiguous b128 LDS loads.
// B frag (32x16 f16, N across lanes): lane l: col=l&15, k same as A frag
//   -> weights stored transposed [Hpad][Kpad] -> two contiguous b128 loads.
// C/D f32: row = v + (l>>4)*8, col = l&15.
__device__ __forceinline__ v8f wmma_tile(const _Float16* __restrict__ src, int kp,
                                         const _Float16* __restrict__ wt,
                                         int m0, int n0, int lane) {
  v8f c = {};
  const int l15 = lane & 15, hh = lane >> 4;
  const _Float16* ap = src + (m0 + l15) * kp + hh * 8;
  const _Float16* bp = wt + (long long)(n0 + l15) * kp + hh * 8;
  for (int kb = 0; kb < kp; kb += 32) {
    V16U a, b;
    a.h[0] = *(const v8h*)(ap + kb);
    a.h[1] = *(const v8h*)(ap + kb + 16);
    b.h[0] = *(const v8h*)(bp + kb);
    b.h[1] = *(const v8h*)(bp + kb + 16);
    c = __builtin_amdgcn_wmma_f32_16x16x32_f16(false, a.v, false, b.v, (short)0, c, false, false);
  }
  return c;
}

__device__ __forceinline__ void mlp_layer(const _Float16* src, int kp,
                                          const _Float16* __restrict__ wt,
                                          const float* __restrict__ bias,
                                          int H, _Float16* dst, int wave, int lane) {
  const int ntiles = H >> 4;
  for (int t = wave; t < 2 * ntiles; t += 4) {
    int m0 = (t / ntiles) << 4;
    int n0 = (t % ntiles) << 4;
    v8f c = wmma_tile(src, kp, wt, m0, n0, lane);
    float bv = bias[n0 + (lane & 15)];
    int hh = lane >> 4;
#pragma unroll
    for (int v2 = 0; v2 < 8; ++v2) {
      float val = fmaxf(c[v2] + bv, 0.0f);                       // ReLU (plain_last=False)
      dst[(m0 + hh * 8 + v2) * H + n0 + (lane & 15)] = (_Float16)val;
    }
  }
}

__global__ void __launch_bounds__(128)
mlp_kernel(const _Float16* __restrict__ in, int kpad0, int sAelems,
           const _Float16* __restrict__ wt1, const float* __restrict__ b1, int h1,
           const _Float16* __restrict__ wt2, const float* __restrict__ b2, int h2,
           const _Float16* __restrict__ wt3, const float* __restrict__ b3, int h3,
           int reluLast, int aggregate, const unsigned* __restrict__ masks,
           float* __restrict__ out, int ldOut, int realH) {
  extern __shared__ char smemRaw[];
  _Float16* sA   = (_Float16*)smemRaw;
  _Float16* sB   = (_Float16*)(smemRaw + (size_t)sAelems * 2);
  float*    sAgg = (float*)(smemRaw + (size_t)sAelems * 2 + 32 * 256 * 2);

  const int tid = threadIdx.x, lane = tid & 31, wave = tid >> 5;
  const long long rowBase = (long long)blockIdx.x * 32;

  { // stage 32 input rows into LDS (b128 chunks)
    const v8h* gin = (const v8h*)(in + rowBase * kpad0);
    v8h* dst = (v8h*)sA;
    int nch = (32 * kpad0) >> 3;
    for (int i = tid; i < nch; i += 128) dst[i] = gin[i];
  }
  __syncthreads();

  mlp_layer(sA, kpad0, wt1, b1, h1, sB, wave, lane);
  __syncthreads();

  const _Float16* fsrc; const _Float16* fwt; const float* fb; int fkp, H;
  if (h3 > 0) {
    mlp_layer(sB, h1, wt2, b2, h2, sA, wave, lane);
    __syncthreads();
    fsrc = sA; fkp = h2; fwt = wt3; fb = b3; H = h3;
  } else {
    fsrc = sB; fkp = h1; fwt = wt2; fb = b2; H = h2;
  }

  const int ntiles = H >> 4;
  const unsigned vm = aggregate ? masks[blockIdx.x] : 0u;
  for (int t = wave; t < 2 * ntiles; t += 4) {
    int m0 = (t / ntiles) << 4;
    int n0 = (t % ntiles) << 4;
    v8f c = wmma_tile(fsrc, fkp, fwt, m0, n0, lane);
    float bv = fb[n0 + (lane & 15)];
    int hh = lane >> 4;
    if (aggregate) {
      float mx = -1e9f;                                  // == reference NEG fill
#pragma unroll
      for (int v2 = 0; v2 < 8; ++v2) {
        float val = c[v2] + bv;
        if (reluLast) val = fmaxf(val, 0.0f);
        int rbit = m0 + hh * 8 + v2;
        if ((vm >> rbit) & 1u) mx = fmaxf(mx, val);
      }
      float o = __shfl_xor(mx, 16);
      mx = fmaxf(mx, o);
      if (hh == 0) sAgg[(m0 >> 4) * 512 + n0 + lane] = mx;
    } else {
      int col = n0 + (lane & 15);
#pragma unroll
      for (int v2 = 0; v2 < 8; ++v2) {
        float val = c[v2] + bv;
        if (reluLast) val = fmaxf(val, 0.0f);
        long long grow = rowBase + m0 + hh * 8 + v2;
        if (col < realH) out[grow * ldOut + col] = val;
      }
    }
  }
  if (aggregate) {
    __syncthreads();
    for (int c0 = tid; c0 < H; c0 += 128)
      out[(long long)blockIdx.x * ldOut + c0] = fmaxf(sAgg[c0], sAgg[512 + c0]);
  }
}

// ============================================================================
// Host side
// ============================================================================
extern "C" void kernel_launch(void* const* d_in, const int* in_sizes, int n_in,
                              void* d_out, int out_size, void* d_ws, size_t ws_size,
                              hipStream_t stream) {
  (void)in_sizes; (void)n_in; (void)out_size; (void)ws_size;
  const float* x0 = (const float*)d_in[0];
  const float* p0 = (const float*)d_in[1];

  // allow big dynamic LDS for the MLP kernel (fp4 input tile = 768 cols).
  // Idempotent host-side call, executed unconditionally (deterministic).
  (void)hipFuncSetAttribute(reinterpret_cast<const void*>(&mlp_kernel),
                            hipFuncAttributeMaxDynamicSharedMemorySize, 160 * 1024);

  char* ws = (char*)d_ws; size_t off = 0;
  auto alloc = [&](size_t bytes) -> void* {
    off = (off + 255) & ~size_t(255);
    void* p = ws + off; off += bytes; return p;
  };

  // ---- weight prep: 24 layers, d_in[3 + 2*i] = W, [4 + 2*i] = b ----
  static const int LK[24] = {  6, 32, 32,  67, 64, 64, 131,128,128, 259,256,256,
                              768,256, 384,256, 320,256, 131,128,128, 128,128,128 };
  static const int LH[24] = { 32, 32, 64,  64, 64,128, 128,128,256, 256,256,512,
                              256,256, 256,256, 256,128, 128,128,128, 128,128, 13 };
  const _Float16* WT[24]; const float* BP[24]; int HP[24];
  for (int i = 0; i < 24; ++i) {
    int K = LK[i], H = LH[i];
    int kp = (K + 31) & ~31, hp = (H + 15) & ~15;
    _Float16* wt = (_Float16*)alloc((size_t)kp * hp * 2);
    float*    bp = (float*)alloc((size_t)hp * 4);
    const float* W = (const float*)d_in[3 + 2 * i];
    const float* b = (const float*)d_in[4 + 2 * i];
    int total = kp * hp;
    prep_weight_kernel<<<(total + 255) / 256, 256, 0, stream>>>(W, b, K, H, kp, hp, wt, bp);
    WT[i] = wt; BP[i] = bp; HP[i] = hp;
  }

  auto launch_mlp = [&](const _Float16* in, int nRows, int kpad0,
                        int l1, int l2, int l3, int reluLast, int aggregate,
                        const unsigned* masks, float* out, int ldOut, int realH) {
    int h1 = HP[l1], h2 = HP[l2], h3 = (l3 >= 0) ? HP[l3] : 0;
    int sAelems = 32 * kpad0;
    if (l3 >= 0 && 32 * h2 > sAelems) sAelems = 32 * h2;
    size_t shmem = (size_t)sAelems * 2 + 32 * 256 * 2 + 2 * 512 * 4;
    mlp_kernel<<<nRows / 32, 128, shmem, stream>>>(
        in, kpad0, sAelems,
        WT[l1], BP[l1], h1, WT[l2], BP[l2], h2,
        (l3 >= 0) ? WT[l3] : (const _Float16*)nullptr,
        (l3 >= 0) ? BP[l3] : (const float*)nullptr, h3,
        reluLast, aggregate, masks, out, ldOut, realH);
  };

  // ---- workspace ----
  int* idx1 = (int*)alloc(4096 * 4);  int* idx2 = (int*)alloc(1024 * 4);
  int* idx3 = (int*)alloc(256 * 4);   int* idx4 = (int*)alloc(64 * 4);
  float* p1 = (float*)alloc(4096 * 3 * 4); float* p2 = (float*)alloc(1024 * 3 * 4);
  float* p3 = (float*)alloc(256 * 3 * 4);  float* p4 = (float*)alloc(64 * 3 * 4);
  int* nbr1 = (int*)alloc((size_t)4096 * 32 * 4); unsigned* m1 = (unsigned*)alloc(4096 * 4);
  int* nbr2 = (int*)alloc((size_t)1024 * 32 * 4); unsigned* m2 = (unsigned*)alloc(1024 * 4);
  int* nbr3 = (int*)alloc((size_t)256 * 32 * 4);  unsigned* m3 = (unsigned*)alloc(256 * 4);
  int* nbr4 = (int*)alloc((size_t)64 * 32 * 4);   unsigned* m4 = (unsigned*)alloc(64 * 4);
  _Float16* edgeIn = (_Float16*)alloc((size_t)131072 * 32 * 2);   // reused, max region
  float* x1 = (float*)alloc((size_t)4096 * 64 * 4);
  float* x2 = (float*)alloc((size_t)1024 * 128 * 4);
  float* x3 = (float*)alloc((size_t)256 * 256 * 4);
  float* x4 = (float*)alloc((size_t)64 * 512 * 4);
  int*   knnIdx = (int*)alloc((size_t)16384 * 3 * 4);
  float* knnW   = (float*)alloc((size_t)16384 * 3 * 4);
  float* interpBuf = (float*)alloc((size_t)16384 * 128 * 4);      // reused, max region
  float* g4 = (float*)alloc((size_t)256 * 256 * 4);
  float* g3 = (float*)alloc((size_t)1024 * 256 * 4);
  float* g2 = (float*)alloc((size_t)4096 * 128 * 4);
  float* g1 = (float*)alloc((size_t)16384 * 128 * 4);

  // ---- SA1: 16384 -> 4096, r=0.1, feat 3 -> [6]->32->32->64 ----
  fps_kernel<<<1, 1024, 0, stream>>>(p0, 16384, 4096, idx1);
  gather_pos_kernel<<<(4096 * 3 + 255) / 256, 256, 0, stream>>>(p0, idx1, 4096, p1);
  ball_query_kernel<<<(4096 + 127) / 128, 128, 0, stream>>>(p1, 4096, p0, 16384, 0.01f, nbr1, m1);
  build_sa_input_kernel<<<(int)(((size_t)131072 * 32 + 255) / 256), 256, 0, stream>>>(
      x0, 3, p0, p1, nbr1, 131072, 32, edgeIn);
  launch_mlp(edgeIn, 131072, 32, 0, 1, 2, 1, 1, m1, x1, 64, 64);

  // ---- SA2: 4096 -> 1024, r=0.2, [67]->64->64->128 ----
  fps_kernel<<<1, 1024, 0, stream>>>(p1, 4096, 1024, idx2);
  gather_pos_kernel<<<(1024 * 3 + 255) / 256, 256, 0, stream>>>(p1, idx2, 1024, p2);
  ball_query_kernel<<<(1024 + 127) / 128, 128, 0, stream>>>(p2, 1024, p1, 4096, 0.04f, nbr2, m2);
  build_sa_input_kernel<<<(int)(((size_t)32768 * 96 + 255) / 256), 256, 0, stream>>>(
      x1, 64, p1, p2, nbr2, 32768, 96, edgeIn);
  launch_mlp(edgeIn, 32768, 96, 3, 4, 5, 1, 1, m2, x2, 128, 128);

  // ---- SA3: 1024 -> 256, r=0.4, [131]->128->128->256 ----
  fps_kernel<<<1, 1024, 0, stream>>>(p2, 1024, 256, idx3);
  gather_pos_kernel<<<(256 * 3 + 255) / 256, 256, 0, stream>>>(p2, idx3, 256, p3);
  ball_query_kernel<<<(256 + 127) / 128, 128, 0, stream>>>(p3, 256, p2, 1024, 0.16f, nbr3, m3);
  build_sa_input_kernel<<<(int)(((size_t)8192 * 160 + 255) / 256), 256, 0, stream>>>(
      x2, 128, p2, p3, nbr3, 8192, 160, edgeIn);
  launch_mlp(edgeIn, 8192, 160, 6, 7, 8, 1, 1, m3, x3, 256, 256);

  // ---- SA4: 256 -> 64, r=0.8, [259]->256->256->512 ----
  fps_kernel<<<1, 1024, 0, stream>>>(p3, 256, 64, idx4);
  gather_pos_kernel<<<(64 * 3 + 255) / 256, 256, 0, stream>>>(p3, idx4, 64, p4);
  ball_query_kernel<<<1, 128, 0, stream>>>(p4, 64, p3, 256, 0.64f, nbr4, m4);
  build_sa_input_kernel<<<(int)(((size_t)2048 * 288 + 255) / 256), 256, 0, stream>>>(
      x3, 256, p3, p4, nbr4, 2048, 288, edgeIn);
  launch_mlp(edgeIn, 2048, 288, 9, 10, 11, 1, 1, m4, x4, 512, 512);

  // ---- FP4: interp x4(p4)->p3, concat x3: [768]->256->256 ----
  knn3_kernel<<<(256 + 127) / 128, 128, 0, stream>>>(p3, 256, p4, 64, knnIdx, knnW);
  interp_kernel<<<(256 * 512 + 255) / 256, 256, 0, stream>>>(x4, 512, knnIdx, knnW, 256, interpBuf);
  build_fp_input_kernel<<<(int)(((size_t)256 * 768 + 255) / 256), 256, 0, stream>>>(
      interpBuf, 512, x3, 256, 256, 768, edgeIn);
  launch_mlp(edgeIn, 256, 768, 12, 13, -1, 1, 0, nullptr, g4, 256, 256);

  // ---- FP3: interp g4(p3)->p2, concat x2: [384]->256->256 ----
  knn3_kernel<<<(1024 + 127) / 128, 128, 0, stream>>>(p2, 1024, p3, 256, knnIdx, knnW);
  interp_kernel<<<(1024 * 256 + 255) / 256, 256, 0, stream>>>(g4, 256, knnIdx, knnW, 1024, interpBuf);
  build_fp_input_kernel<<<(int)(((size_t)1024 * 384 + 255) / 256), 256, 0, stream>>>(
      interpBuf, 256, x2, 128, 1024, 384, edgeIn);
  launch_mlp(edgeIn, 1024, 384, 14, 15, -1, 1, 0, nullptr, g3, 256, 256);

  // ---- FP2: interp g3(p2)->p1, concat x1: [320]->256->128 ----
  knn3_kernel<<<(4096 + 127) / 128, 128, 0, stream>>>(p1, 4096, p2, 1024, knnIdx, knnW);
  interp_kernel<<<(int)(((size_t)4096 * 256 + 255) / 256), 256, 0, stream>>>(
      g3, 256, knnIdx, knnW, 4096, interpBuf);
  build_fp_input_kernel<<<(int)(((size_t)4096 * 320 + 255) / 256), 256, 0, stream>>>(
      interpBuf, 256, x1, 64, 4096, 320, edgeIn);
  launch_mlp(edgeIn, 4096, 320, 16, 17, -1, 1, 0, nullptr, g2, 128, 128);

  // ---- FP1: interp g2(p1)->p0, concat x0: [131]->128->128->128 ----
  knn3_kernel<<<(16384 + 127) / 128, 128, 0, stream>>>(p0, 16384, p1, 4096, knnIdx, knnW);
  interp_kernel<<<(int)(((size_t)16384 * 128 + 255) / 256), 256, 0, stream>>>(
      g2, 128, knnIdx, knnW, 16384, interpBuf);
  build_fp_input_kernel<<<(int)(((size_t)16384 * 160 + 255) / 256), 256, 0, stream>>>(
      interpBuf, 128, x0, 3, 16384, 160, edgeIn);
  launch_mlp(edgeIn, 16384, 160, 18, 19, 20, 1, 0, nullptr, g1, 128, 128);

  // ---- head: 128->128->128->13 (final plain linear) ----
  build_fp_input_kernel<<<(int)(((size_t)16384 * 128 + 255) / 256), 256, 0, stream>>>(
      g1, 128, nullptr, 0, 16384, 128, edgeIn);
  launch_mlp(edgeIn, 16384, 128, 21, 22, 23, 0, 0, nullptr, (float*)d_out, 13, 13);
}
